// SpatialMHSA_72791105733232
// MI455X (gfx1250) — matrix operand order repaired and checked by reading, not compiled, not run on hardware
//
#include <hip/hip_runtime.h>
#include <hip/hip_bf16.h>

// SpatialMHSA for MI455X (gfx1250): bf16 WMMA pipeline + TDM async K/V staging.
// B=4 T=12 N=512 D=256 H=8 hd=32.  hd==32 => one v_wmma_f32_16x16x32_bf16 per
// 16x16 score tile (full K in a single WMMA).

#define BB 4
#define TT 12
#define NN 512
#define DD 256
#define HH 8
#define HD 32
#define BT (BB * TT)      // 48
#define MTOT (BT * NN)    // 24576 rows in the projection GEMMs

typedef __attribute__((ext_vector_type(16))) __bf16 bf16x16;
typedef __attribute__((ext_vector_type(8)))  float  f32x8;
typedef __attribute__((ext_vector_type(4))) unsigned int u32x4;
typedef __attribute__((ext_vector_type(8)))  int i32x8;
typedef __attribute__((ext_vector_type(4)))  int i32x4;

union FragB {
    uint4 u[2];
    bf16x16 v;
};

__device__ __forceinline__ f32x8 wmma_bf16(bf16x16 a, bf16x16 b, f32x8 c) {
    return __builtin_amdgcn_wmma_f32_16x16x32_bf16(
        /*neg_a=*/false, a, /*neg_b=*/false, b,
        /*c_mod=*/(short)0, c, /*reuse_a=*/false, /*reuse_b=*/false);
}

__device__ __forceinline__ void cvt8(bf16x16& af, int base, float4 a, float4 b) {
    af[base + 0] = (__bf16)a.x; af[base + 1] = (__bf16)a.y;
    af[base + 2] = (__bf16)a.z; af[base + 3] = (__bf16)a.w;
    af[base + 4] = (__bf16)b.x; af[base + 5] = (__bf16)b.y;
    af[base + 6] = (__bf16)b.z; af[base + 7] = (__bf16)b.w;
}

// 1-D Tensor Data Mover copy: nelem bf16 elements, global -> LDS.
// D# layout per CDNA5 ISA:
//   group0: [1:0]=count=1, [63:32]=lds_addr, [120:64]=global_addr, [127:126]=type=2
//   group1: [17:16]=data_size=1 (2B), [79:48]=tensor_dim0, [111:80]=tensor_dim1=1,
//           [127:112]=tile_dim0, [207:160]=tensor_dim0_stride
// Toolchain here uses the 6-arg builtin (clang-23 / therock headers):
//   (u32x4 g0, i32x8 g1, i32x4 g2, i32x4 g3, i32x8 pad, i32 cpol)
__device__ __forceinline__ void tdm_load_1d(unsigned int lds_off, const void* gptr,
                                            unsigned int nelem) {
    unsigned long long ga = (unsigned long long)(size_t)gptr;
    u32x4 g0;
    g0[0] = 1u;                                                  // count=1
    g0[1] = lds_off;                                             // lds byte addr
    g0[2] = (unsigned int)ga;                                    // global_addr lo
    g0[3] = ((unsigned int)(ga >> 32) & 0x01FFFFFFu) | 0x80000000u;  // hi | type=2
    i32x8 g1;
    g1[0] = 0x10000;                                             // data_size = 2B
    g1[1] = (int)((nelem & 0xFFFFu) << 16);                      // tensor_dim0 lo16
    g1[2] = (int)(((nelem >> 16) & 0xFFFFu) | (1u << 16));       // dim0 hi | dim1=1
    g1[3] = (int)((nelem & 0xFFFFu) << 16);                      // tile_dim0
    g1[4] = 0;                                                   // tile_dim1/2 unused
    g1[5] = (int)nelem;                                          // dim0_stride lo
    g1[6] = 0;
    g1[7] = 0;
    i32x4 z4 = {0, 0, 0, 0};
    i32x8 z8 = {0, 0, 0, 0, 0, 0, 0, 0};
    __builtin_amdgcn_tensor_load_to_lds(g0, g1, z4, z4, z8, 0);
}

// ---------------------------------------------------------------------------
// Kernel 0: transpose+convert the 4 weight matrices to bf16 (Wt[n][k]=W[k][n])
// so WMMA B-fragments become contiguous per-lane loads.
// ---------------------------------------------------------------------------
__global__ __launch_bounds__(256) void prep_weights(
    const float* __restrict__ Wq, const float* __restrict__ Wk,
    const float* __restrict__ Wv, const float* __restrict__ Wo,
    __bf16* __restrict__ wt) {
    const int n = threadIdx.x;   // column of original W
    const int k = blockIdx.x;    // row of original W
    const int m = blockIdx.y;    // which matrix
    const float* W = (m == 0) ? Wq : (m == 1) ? Wk : (m == 2) ? Wv : Wo;
    wt[(size_t)m * DD * DD + (size_t)n * DD + k] = (__bf16)W[(size_t)k * DD + n];
}

// ---------------------------------------------------------------------------
// Kernel 1: QKV projection.  Each wave computes a 16x64 output tile (4 C tiles,
// 32 accum VGPRs), k-loop over 256 in steps of 32.  Q,K stored [bth][n][hd];
// V stored transposed [bth][hd][n] so the attention PV B-fragments are
// contiguous.
// ---------------------------------------------------------------------------
__global__ __launch_bounds__(256) void qkv_proj(
    const float* __restrict__ x, const __bf16* __restrict__ wt,
    __bf16* __restrict__ qb, __bf16* __restrict__ kb, __bf16* __restrict__ vt) {
    const int lane = threadIdx.x & 31;
    const int wv   = threadIdx.x >> 5;
    const int which = blockIdx.z;                    // 0=q 1=k 2=v
    const int r0 = (blockIdx.x * 8 + wv) * 16;       // row tile base
    const int c0 = blockIdx.y * 64;                  // column group base
    const __bf16* Bw = wt + (size_t)which * (DD * DD);

    const int la  = lane & 15;
    const int aHi = (lane & 16) ? 8 : 0;             // A-frag K phase
    const int bHi = (lane & 16) ? 8 : 0;             // B-frag K phase

    f32x8 acc[4] = {};
    const float* xrow = x + (size_t)(r0 + la) * DD;

    for (int kk = 0; kk < DD; kk += 32) {
        // A fragment: x row, fp32 -> bf16
        const float* xa = xrow + kk + aHi;
        float4 a0 = *(const float4*)(xa);
        float4 a1 = *(const float4*)(xa + 4);
        float4 a2 = *(const float4*)(xa + 16);
        float4 a3 = *(const float4*)(xa + 20);
        bf16x16 af;
        cvt8(af, 0, a0, a1);
        cvt8(af, 8, a2, a3);
#pragma unroll
        for (int t = 0; t < 4; ++t) {
            const __bf16* bp = Bw + (size_t)(c0 + t * 16 + la) * DD + kk + bHi;
            FragB fb;
            fb.u[0] = *(const uint4*)bp;
            fb.u[1] = *(const uint4*)(bp + 16);
            acc[t] = wmma_bf16(af, fb.v, acc[t]);
        }
    }

    // C layout: lane holds column (lane&15); VGPR j holds row j (+8 if lane>=16)
    const int hi = aHi;  // same 0/8 split
    const int bt = r0 >> 9;        // 16-row tiles never straddle a 512-row block
    const int nbase = r0 & 511;
    if (which == 2) {
#pragma unroll
        for (int t = 0; t < 4; ++t) {
            const int d = c0 + t * 16 + la;
            const int h = d >> 5, dh = d & 31;
            __bf16* vrow = vt + (((size_t)bt * HH + h) * HD + dh) * NN + nbase + hi;
#pragma unroll
            for (int j = 0; j < 8; ++j)
                vrow[j] = (__bf16)acc[t][j];
        }
    } else {
        __bf16* dst = which ? kb : qb;
#pragma unroll
        for (int t = 0; t < 4; ++t) {
            const int d = c0 + t * 16 + la;
            const int h = d >> 5, dh = d & 31;
            __bf16* drow = dst + (((size_t)bt * HH + h) * NN + nbase + hi) * HD + dh;
#pragma unroll
            for (int j = 0; j < 8; ++j)
                drow[(size_t)j * HD] = (__bf16)acc[t][j];
        }
    }
}

// ---------------------------------------------------------------------------
// Kernel 2: attention.  One workgroup = one (b,t,h) x 64-query block.
// TDM DMAs this head's K (512x32) and V^T (32x512) blocks into LDS up front
// (tensor_load_to_lds + s_wait_tensorcnt).  4 waves x 16 query rows.
// Pass1: S = Q K^T (one WMMA per tile, K frags via ds_load) -> LDS fp32.
// Softmax (scale + sim bias) thread-per-row in fp32, P -> LDS bf16.
// Pass2: out = P V via WMMA, A- and B-frags both from LDS.
// LDS: 128KB scores + 64KB probs + 32KB K + 32KB V = 256KB (<320KB/WGP).
// ---------------------------------------------------------------------------
__global__ __launch_bounds__(128) void attn(
    const __bf16* __restrict__ qb, const __bf16* __restrict__ kb,
    const __bf16* __restrict__ vt, const float* __restrict__ sim,
    __bf16* __restrict__ ctx) {
    __shared__ float  S[64 * 512];     // raw scores, fp32 (128KB)
    __shared__ __bf16 P[64 * 512];     // normalized probs, bf16 (64KB)
    __shared__ __bf16 KL[NN * HD];     // K block   [n][hd]   (32KB)
    __shared__ __bf16 VL[HD * NN];     // V^T block [hd][n]   (32KB)

    const int lane = threadIdx.x & 31;
    const int wv   = threadIdx.x >> 5;          // 0..3
    const int bth  = blockIdx.x;                // bt*H + h
    const int q0   = blockIdx.y * 64;           // query block base
    const int qr0  = q0 + wv * 16;              // this wave's 16-row strip
    const int la   = lane & 15;
    const int aHi  = (lane & 16) ? 8 : 0;
    const int bHi  = (lane & 16) ? 8 : 0;
    const int hi   = aHi;

    // ---- TDM: stage K and V^T for this head into LDS (wave 0 issues DMA) ----
    if (threadIdx.x < 32) {
        const unsigned int koff = (unsigned int)(size_t)(void*)KL;
        const unsigned int voff = (unsigned int)(size_t)(void*)VL;
        tdm_load_1d(koff, kb + (size_t)bth * NN * HD, NN * HD);
        tdm_load_1d(voff, vt + (size_t)bth * HD * NN, HD * NN);
        __builtin_amdgcn_s_wait_tensorcnt(0);
    }
    __syncthreads();

    // Q A-fragment: reused across all 32 key tiles
    const __bf16* Qp = qb + ((size_t)bth * NN + qr0 + la) * HD + aHi;
    FragB qf;
    qf.u[0] = *(const uint4*)Qp;
    qf.u[1] = *(const uint4*)(Qp + 16);

    float* Srow = &S[(size_t)(wv * 16) * 512];

    // ---- pass 1: scores -> LDS ----
    for (int mt = 0; mt < 32; ++mt) {
        const int m0 = mt * 16;
        const __bf16* Kp = KL + (size_t)(m0 + la) * HD + bHi;   // ds_load_b128 x2
        FragB kf;
        kf.u[0] = *(const uint4*)Kp;
        kf.u[1] = *(const uint4*)(Kp + 16);
        f32x8 c = {};
        c = wmma_bf16(qf.v, kf.v, c);
#pragma unroll
        for (int j = 0; j < 8; ++j)
            Srow[(hi + j) * 512 + m0 + la] = c[j];
    }
    __syncthreads();

    // ---- softmax: one thread per query row (threads 0..63) ----
    if (threadIdx.x < 64) {
        float* row = &S[(size_t)threadIdx.x * 512];
        __bf16* prow = &P[(size_t)threadIdx.x * 512];
        const float* simrow = sim + (size_t)(q0 + threadIdx.x) * NN;
        const float scale = 0.17677669529663687f;  // 1/sqrt(32)
        float mx = -3.402823466e38f;
        for (int m = 0; m < NN; ++m) {
            const float v = row[m] * scale + simrow[m];
            row[m] = v;
            mx = fmaxf(mx, v);
        }
        float sum = 0.f;
        for (int m = 0; m < NN; ++m) {
            const float e = __expf(row[m] - mx);
            row[m] = e;
            sum += e;
        }
        const float inv = 1.f / sum;
        for (int m = 0; m < NN; ++m)
            prow[m] = (__bf16)(row[m] * inv);
    }
    __syncthreads();

    // ---- pass 2: out = P @ V  (two 16x16 output tiles = hd 32) ----
    f32x8 o0 = {}, o1 = {};
    const __bf16* Pbase  = &P[(size_t)(wv * 16 + la) * 512];
    const __bf16* v0base = VL + (size_t)(0  + la) * NN;
    const __bf16* v1base = VL + (size_t)(16 + la) * NN;
    for (int ms = 0; ms < 16; ++ms) {
        const int mb = ms * 32;
        const __bf16* pa = Pbase + mb + aHi;
        FragB pf;
        pf.u[0] = *(const uint4*)pa;          // ds_load_b128
        pf.u[1] = *(const uint4*)(pa + 16);
        const __bf16* vp0 = v0base + mb + bHi;
        const __bf16* vp1 = v1base + mb + bHi;
        FragB f0, f1;
        f0.u[0] = *(const uint4*)vp0; f0.u[1] = *(const uint4*)(vp0 + 16);
        f1.u[0] = *(const uint4*)vp1; f1.u[1] = *(const uint4*)(vp1 + 16);
        o0 = wmma_bf16(pf.v, f0.v, o0);
        o1 = wmma_bf16(pf.v, f1.v, o1);
    }

    // merge heads: ctx[bt][n][h*32 + d]
    const int h  = bth % HH;
    const int bt = bth / HH;
#pragma unroll
    for (int j = 0; j < 8; ++j) {
        const int n = qr0 + hi + j;
        const size_t base = ((size_t)bt * NN + n) * DD + h * HD;
        ctx[base + la]      = (__bf16)o0[j];
        ctx[base + 16 + la] = (__bf16)o1[j];
    }
}

// ---------------------------------------------------------------------------
// Kernel 3: output projection ctx @ Wo -> fp32 out.  Same tiling as kernel 1.
// ---------------------------------------------------------------------------
__global__ __launch_bounds__(256) void out_proj(
    const __bf16* __restrict__ ctx, const __bf16* __restrict__ wto,
    float* __restrict__ out) {
    const int lane = threadIdx.x & 31;
    const int wv   = threadIdx.x >> 5;
    const int r0 = (blockIdx.x * 8 + wv) * 16;
    const int c0 = blockIdx.y * 64;
    const int la  = lane & 15;
    const int aHi = (lane & 16) ? 8 : 0;
    const int bHi = (lane & 16) ? 8 : 0;

    f32x8 acc[4] = {};
    const __bf16* arow = ctx + (size_t)(r0 + la) * DD;

    for (int kk = 0; kk < DD; kk += 32) {
        const __bf16* ap = arow + kk + aHi;
        FragB af;
        af.u[0] = *(const uint4*)ap;
        af.u[1] = *(const uint4*)(ap + 16);
#pragma unroll
        for (int t = 0; t < 4; ++t) {
            const __bf16* bp = wto + (size_t)(c0 + t * 16 + la) * DD + kk + bHi;
            FragB fb;
            fb.u[0] = *(const uint4*)bp;
            fb.u[1] = *(const uint4*)(bp + 16);
            acc[t] = wmma_bf16(af.v, fb.v, acc[t]);
        }
    }

    const int hi = aHi;
#pragma unroll
    for (int t = 0; t < 4; ++t) {
#pragma unroll
        for (int j = 0; j < 8; ++j)
            out[(size_t)(r0 + hi + j) * DD + c0 + t * 16 + la] = acc[t][j];
    }
}

// ---------------------------------------------------------------------------
extern "C" void kernel_launch(void* const* d_in, const int* in_sizes, int n_in,
                              void* d_out, int out_size, void* d_ws, size_t ws_size,
                              hipStream_t stream) {
    const float* x   = (const float*)d_in[0];
    const float* sim = (const float*)d_in[1];
    const float* Wq  = (const float*)d_in[2];
    const float* Wk  = (const float*)d_in[3];
    const float* Wv  = (const float*)d_in[4];
    const float* Wo  = (const float*)d_in[5];
    float* out = (float*)d_out;

    // workspace carve-up (bf16 elements); total ~48.5 MB
    __bf16* wt  = (__bf16*)d_ws;                       // 4 x 256x256 transposed weights
    __bf16* qb  = wt + (size_t)4 * DD * DD;            // [BT,H,N,hd]
    __bf16* kb  = qb + (size_t)MTOT * DD;              // [BT,H,N,hd]
    __bf16* vt  = kb + (size_t)MTOT * DD;              // [BT,H,hd,N] (transposed)
    __bf16* ctx = vt + (size_t)MTOT * DD;              // [BT,N,D]

    prep_weights<<<dim3(DD, 4), 256, 0, stream>>>(Wq, Wk, Wv, Wo, wt);
    qkv_proj<<<dim3(MTOT / 16 / 8, 4, 3), 256, 0, stream>>>(x, wt, qb, kb, vt);
    attn<<<dim3(BT * HH, NN / 64), 128, 0, stream>>>(qb, kb, vt, sim, ctx);
    out_proj<<<dim3(MTOT / 16 / 8, 4), 256, 0, stream>>>(ctx, wt + (size_t)3 * DD * DD, out);
}
